// AnsweringHead_28286654611509
// MI455X (gfx1250) — matrix operand order repaired and checked by reading, not compiled, not run on hardware
//
#include <hip/hip_runtime.h>
#include <hip/hip_bf16.h>
#include <math.h>

typedef __attribute__((ext_vector_type(16))) __bf16 v16bf;
typedef __attribute__((ext_vector_type(8)))  float  v8f;

#define BB    2
#define SS    64
#define DD    768
#define VV    30522
#define ROWS  (BB * SS)        // 128
#define KSEL  256
#define EPSV  1.17549435e-38f  // np.float32 tiny
#define NEGINF (-__builtin_inff())

// ---------------- reduction helpers (wave32) ----------------
__device__ __forceinline__ float waveMax(float v) {
#pragma unroll
  for (int off = 16; off > 0; off >>= 1)
    v = fmaxf(v, __shfl_down(v, off, 32));
  return v;
}
__device__ __forceinline__ float waveSum(float v) {
#pragma unroll
  for (int off = 16; off > 0; off >>= 1)
    v += __shfl_down(v, off, 32);
  return v;
}
// red must have >= 33 floats of LDS
__device__ __forceinline__ float blockMax(float v, float* red) {
  const int lid = threadIdx.x & 31, wid = threadIdx.x >> 5, nw = blockDim.x >> 5;
  v = waveMax(v);
  if (lid == 0) red[wid] = v;
  __syncthreads();
  if (wid == 0) {
    float x = (lid < nw) ? red[lid] : NEGINF;
    x = waveMax(x);
    if (lid == 0) red[32] = x;
  }
  __syncthreads();
  float r = red[32];
  __syncthreads();
  return r;
}
__device__ __forceinline__ float blockSum(float v, float* red) {
  const int lid = threadIdx.x & 31, wid = threadIdx.x >> 5, nw = blockDim.x >> 5;
  v = waveSum(v);
  if (lid == 0) red[wid] = v;
  __syncthreads();
  if (wid == 0) {
    float x = (lid < nw) ? red[lid] : 0.0f;
    x = waveSum(x);
    if (lid == 0) red[32] = x;
  }
  __syncthreads();
  float r = red[32];
  __syncthreads();
  return r;
}

// ---------------- 1) logits GEMM via bf16 WMMA ----------------
// logits[r, n] = sum_k embed[ids[r], k] * W[k, n] + bias[n]
// One wave owns a full 128-row M strip for one 16-column N tile:
// per K-step (32), the B fragment (strided W loads — the only HBM-heavy
// operand) is loaded ONCE and reused by 8 WMMAs against 8 accumulators.
// W is then read exactly once from HBM (93.8 MB -> ~4 us at 23.3 TB/s);
// A (393 KB total) stays cache-hot.
__global__ __launch_bounds__(128)
void gemm_logits_kernel(const int* __restrict__ input_ids,
                        const float* __restrict__ embed,   // (V, D)
                        const float* __restrict__ W,       // (D, V)
                        const float* __restrict__ bias,    // (V)
                        float* __restrict__ logits)        // (ROWS, V)
{
  const int nTilesN = (VV + 15) / 16;          // 1908
  const int wave = blockIdx.x * (blockDim.x >> 5) + (threadIdx.x >> 5);
  if (wave >= nTilesN) return;                 // wave-uniform: EXEC stays all-ones
  const int tileN  = wave;
  const int lane   = threadIdx.x & 31;
  const int laneLo = lane & 15;
  const int hi     = lane >> 4;                // half-wave selector

  // A fragment sources: rows laneLo, 16+laneLo, ..., 112+laneLo of h = embed[token]
  const float* arow[8];
#pragma unroll
  for (int tm = 0; tm < 8; ++tm) {
    const int tok = input_ids[tm * 16 + laneLo];
    arow[tm] = embed + (size_t)tok * DD;
  }

  // B fragment source: column n of W
  const int n  = tileN * 16 + laneLo;
  const int nc = (n < VV) ? n : (VV - 1);      // clamp loads on ragged edge

  v8f acc[8];
#pragma unroll
  for (int tm = 0; tm < 8; ++tm) acc[tm] = (v8f){};

  for (int k0 = 0; k0 < DD; k0 += 32) {
    // B 32x16, contiguous k order shared with A: lanes 0-15 hold K = 0..15, lanes 16-31 hold K = 16..31
    v16bf b;
#pragma unroll
    for (int j = 0; j < 16; ++j) {
      b[j] = (__bf16)W[(size_t)(k0 + hi * 16 + j) * VV + nc];
    }
    // 8 M tiles share this B fragment
#pragma unroll
    for (int tm = 0; tm < 8; ++tm) {
      // A 16x32, contiguous k order shared with B: lanes 0-15 hold K = 0..15, lanes 16-31 hold K = 16..31
      v16bf a;
#pragma unroll
      for (int j = 0; j < 16; ++j) {
        a[j] = (__bf16)arow[tm][k0 + hi * 16 + j];
      }
      acc[tm] = __builtin_amdgcn_wmma_f32_16x16x32_bf16(false, a, false, b,
                                                        (short)0, acc[tm], false, false);
    }
  }
  // f32 C/D 16x16: VGPR i -> row (i + 8*hi), col = laneLo
  if (n < VV) {
    const float bv = bias[n];
#pragma unroll
    for (int tm = 0; tm < 8; ++tm) {
#pragma unroll
      for (int i = 0; i < 8; ++i) {
        const int r = tm * 16 + hi * 8 + i;
        logits[(size_t)r * VV + n] = acc[tm][i] + bv;
      }
    }
  }
}

// ---------------- 2) per-row log-sum-exp of logits ----------------
__global__ __launch_bounds__(256)
void rowstats_kernel(const float* __restrict__ logits, float* __restrict__ lse)
{
  __shared__ float red[33];
  const int row = blockIdx.x;
  const float* lr = logits + (size_t)row * VV;
  float m = NEGINF;
  for (int i = threadIdx.x; i < VV; i += blockDim.x) m = fmaxf(m, lr[i]);
  m = blockMax(m, red);
  float z = 0.0f;
  for (int i = threadIdx.x; i < VV; i += blockDim.x) z += __expf(lr[i] - m);
  z = blockSum(z, red);
  if (threadIdx.x == 0) lse[row] = m + __logf(z);
}

// ---------------- 3) subset operator + hard top-K (1 block = 1 row) ----------------
#define TPB 1024
#define VEC 30  // ceil(30522 / 1024)

__global__ __launch_bounds__(TPB)
void subset_kernel(const float* __restrict__ logits,     // (ROWS, V) in ws
                   const float* __restrict__ gumbel_u,   // (ROWS, V)
                   float* __restrict__ selections)       // (ROWS, V) region of d_out
{
  __shared__ float red[33];
  __shared__ float sAv[32];
  __shared__ int   sAi[32];
  __shared__ int   sWin;

  const int row = blockIdx.x;
  const int tid = threadIdx.x;
  const size_t base = (size_t)row * VV;

  float s[VEC];   // perturbed scores, mutated in place (register-resident)
  float kh[VEC];  // khot accumulator

#pragma unroll
  for (int i = 0; i < VEC; ++i) {
    const int idx = tid + (i << 10);
    kh[i] = 0.0f;
    if (idx < VV) {
      const float u = gumbel_u[base + idx];
      const float g = -__logf(-__logf(u + 1e-20f) + 1e-20f);
      s[i] = logits[base + idx] + g;
    } else {
      s[i] = NEGINF;
    }
  }

  // Xie & Ermon loop, reordered equivalently: p = softmax(s); khot += p; s += log(max(1-p, eps))
  // (original's first s-update is a no-op since onehot starts at zero)
  for (int t = 0; t < KSEL; ++t) {
    float m = NEGINF;
#pragma unroll
    for (int i = 0; i < VEC; ++i) m = fmaxf(m, s[i]);
    m = blockMax(m, red);
    float z = 0.0f;
#pragma unroll
    for (int i = 0; i < VEC; ++i) z += __expf(s[i] - m);
    z = blockSum(z, red);
    const float inv = 1.0f / z;
#pragma unroll
    for (int i = 0; i < VEC; ++i) {
      const float p = __expf(s[i] - m) * inv;
      kh[i] += p;
      s[i]  += __logf(fmaxf(1.0f - p, EPSV));
    }
  }

  // hard top-K on khot: 256 iterative block argmaxes; s becomes the scratch copy
#pragma unroll
  for (int i = 0; i < VEC; ++i) {
    const int idx = tid + (i << 10);
    s[i] = (idx < VV) ? kh[i] : NEGINF;
  }
  unsigned hard = 0u;
  for (int t = 0; t < KSEL; ++t) {
    float bv = NEGINF; int bi = 0x7fffffff;
#pragma unroll
    for (int i = 0; i < VEC; ++i) {
      const int idx = tid + (i << 10);
      if (s[i] > bv) { bv = s[i]; bi = idx; }
    }
#pragma unroll
    for (int off = 16; off > 0; off >>= 1) {
      const float ov = __shfl_down(bv, off, 32);
      const int   oi = __shfl_down(bi, off, 32);
      if (ov > bv || (ov == bv && oi < bi)) { bv = ov; bi = oi; }
    }
    const int lid = tid & 31, wid = tid >> 5;
    if (lid == 0) { sAv[wid] = bv; sAi[wid] = bi; }
    __syncthreads();
    if (wid == 0) {
      float xv = sAv[lid]; int xi = sAi[lid];
#pragma unroll
      for (int off = 16; off > 0; off >>= 1) {
        const float ov = __shfl_down(xv, off, 32);
        const int   oi = __shfl_down(xi, off, 32);
        if (ov > xv || (ov == xv && oi < xi)) { xv = ov; xi = oi; }
      }
      if (lid == 0) sWin = xi;
    }
    __syncthreads();
    const int w = sWin;
    __syncthreads();
    // unrolled compare avoids dynamic register indexing (no scratch spill)
#pragma unroll
    for (int i = 0; i < VEC; ++i) {
      if (w == tid + (i << 10)) { s[i] = NEGINF; hard |= (1u << i); }
    }
  }

  // straight-through in the reference's FP order: hard ? (1 - khot) + khot : (0 - khot) + khot (== 0 exactly)
#pragma unroll
  for (int i = 0; i < VEC; ++i) {
    const int idx = tid + (i << 10);
    if (idx < VV) {
      const float sel = ((hard >> i) & 1u) ? ((1.0f - kh[i]) + kh[i]) : 0.0f;
      selections[base + idx] = sel;
    }
  }
}

// ---------------- 4) reduce over S: values + logprobs ----------------
__global__ __launch_bounds__(256)
void combine_kernel(const float* __restrict__ logits,     // (ROWS, V)
                    const float* __restrict__ lse,        // (ROWS)
                    const float* __restrict__ selections, // (ROWS, V)
                    const float* __restrict__ attn,       // (B, S)
                    float* __restrict__ values,           // (B, V)
                    float* __restrict__ logprobs)         // (B, V)
{
  const int e = blockIdx.x * blockDim.x + threadIdx.x;
  if (e >= BB * VV) return;
  const int b = e / VV;
  const int v = e % VV;
  float lsum = 0.0f;
  float vmax = NEGINF;
  for (int sIdx = 0; sIdx < SS; ++sIdx) {
    const int row = b * SS + sIdx;
    const float lg  = logits[(size_t)row * VV + v];
    const float sel = selections[(size_t)row * VV + v];
    lsum += (lg - lse[row]) * sel;
    const float x = lg * sel;
    const float r = x > 0.0f ? x : 0.0f;       // relu
    const float val = log1pf(r) * attn[row];
    vmax = fmaxf(vmax, val);
  }
  values[e]   = vmax;
  logprobs[e] = lsum;
}

// ---------------- launcher ----------------
extern "C" void kernel_launch(void* const* d_in, const int* in_sizes, int n_in,
                              void* d_out, int out_size, void* d_ws, size_t ws_size,
                              hipStream_t stream) {
  const int*   input_ids = (const int*)  d_in[0];   // (B,S)
  const float* attn      = (const float*)d_in[1];   // (B,S)
  const float* gumbel_u  = (const float*)d_in[2];   // (B*S, V)
  const float* embed     = (const float*)d_in[3];   // (V, D)
  const float* W         = (const float*)d_in[4];   // (D, V)
  const float* bias      = (const float*)d_in[5];   // (V)

  float* out      = (float*)d_out;
  float* values   = out;                          // B*V
  float* logprobs = out + (size_t)BB * VV;        // B*V
  float* sel      = out + (size_t)2 * BB * VV;    // ROWS*V

  // workspace: logits (ROWS*V f32 = 15.63 MB) + per-row LSE (128 f32)
  float* ws_logits = (float*)d_ws;
  float* ws_lse    = ws_logits + (size_t)ROWS * VV;

  const int nTilesN = (VV + 15) / 16;                    // 1908 N tiles, full M strip per wave
  const int gemmBlocks = (nTilesN + 3) / 4;              // 4 waves/block

  gemm_logits_kernel<<<gemmBlocks, 128, 0, stream>>>(input_ids, embed, W, bias, ws_logits);
  rowstats_kernel<<<ROWS, 256, 0, stream>>>(ws_logits, ws_lse);
  subset_kernel<<<ROWS, TPB, 0, stream>>>(ws_logits, gumbel_u, sel);
  combine_kernel<<<(BB * VV + 255) / 256, 256, 0, stream>>>(ws_logits, ws_lse, sel,
                                                            attn, values, logprobs);
}
